// FlashCohereAttention_52415780880574
// MI455X (gfx1250) — matrix operand order repaired and checked by reading, not compiled, not run on hardware
//
#include <hip/hip_runtime.h>
#include <hip/hip_bf16.h>

#define HQ   32
#define HKV  8
#define DH   128
#define HID  4096
#define BB   4
#define SS   1024
#define TT   (BB * SS)
#define NQKV ((HQ + 2 * HKV) * DH) /* 6144 */

typedef __attribute__((ext_vector_type(16))) __bf16       v16bf;
typedef __attribute__((ext_vector_type(8)))  float        v8f;
typedef __attribute__((ext_vector_type(8)))  unsigned int v8u;
typedef __attribute__((ext_vector_type(4)))  int          v4i;
typedef __attribute__((address_space(1))) v4i* gv4i_p;    // global int4*
typedef __attribute__((address_space(3))) v4i* lv4i_p;    // LDS int4*

__device__ __forceinline__ unsigned short f2bf(float f) {
  unsigned int u = __builtin_bit_cast(unsigned int, f);
  u += 0x7fffu + ((u >> 16) & 1u);          // round-to-nearest-even
  return (unsigned short)(u >> 16);
}

// ---- async global->LDS 16B copy (ASYNCcnt path), with safe fallback --------
#if __has_builtin(__builtin_amdgcn_global_load_async_to_lds_b128)
#define ASYNC_OK 1
#endif

__device__ __forceinline__ void async_cp16(const unsigned short* g,
                                           unsigned short* l) {
#ifdef ASYNC_OK
  unsigned short* gnc = const_cast<unsigned short*>(g);
  __builtin_amdgcn_global_load_async_to_lds_b128((gv4i_p)gnc, (lv4i_p)l, 0, 0);
#else
  *(uint4*)l = *(const uint4*)g;
#endif
}

__device__ __forceinline__ void async_wait0() {
#ifdef ASYNC_OK
#if __has_builtin(__builtin_amdgcn_s_wait_asynccnt)
  __builtin_amdgcn_s_wait_asynccnt(0);
#else
  asm volatile("s_wait_asynccnt 0x0" ::: "memory");
#endif
#endif
}

// ---------------------------------------------------------------------------
// fp32 -> bf16 bulk convert (bandwidth-bound, one pass per tensor)
// ---------------------------------------------------------------------------
__global__ __launch_bounds__(256) void cvt_bf16_kernel(
    const float* __restrict__ in, unsigned short* __restrict__ out,
    long long n4) {
  long long i = (long long)blockIdx.x * 256 + threadIdx.x;
  if (i >= n4) return;
  float4 v = ((const float4*)in)[i];
  ushort4 o;
  o.x = f2bf(v.x); o.y = f2bf(v.y); o.z = f2bf(v.z); o.w = f2bf(v.w);
  ((ushort4*)out)[i] = o;
}

// ---------------------------------------------------------------------------
// C[M,N] = A[M,K] * B[N,K]^T  (A,B: bf16, C: fp32). 128x128 tile, BK=32,
// 8 waves x (32x64), double-buffered LDS staged via async global->LDS B128.
// ---------------------------------------------------------------------------
__global__ __launch_bounds__(256) void gemm_bf16_kernel(
    const unsigned short* __restrict__ A, const unsigned short* __restrict__ Bw,
    float* __restrict__ C, int M, int N, int K) {
  const int STR = 40;                       // 80B row stride: 16B-aligned chunks
  __shared__ unsigned short As[2][128 * 40];
  __shared__ unsigned short Bs[2][128 * 40];
  const int tid  = threadIdx.x;
  const int lane = tid & 31;
  const int wave = tid >> 5;
  const int bm = blockIdx.y * 128;
  const int bn = blockIdx.x * 128;
  const int wm = (wave >> 1) * 32;
  const int wn = (wave & 1) * 64;
  const int r  = lane & 15;
  const int hi = lane >> 4;

  v8f acc[2][4];
#pragma unroll
  for (int i = 0; i < 2; ++i)
#pragma unroll
    for (int j = 0; j < 4; ++j) acc[i][j] = (v8f)0.f;

  auto stage = [&](int buf, int kb) {
#pragma unroll
    for (int it = 0; it < 2; ++it) {
      int idx = tid + it * 256;             // 512 B128 chunks per tile
      int row = idx >> 2, c = idx & 3;
      async_cp16(A  + (size_t)(bm + row) * K + kb + c * 8,
                 &As[buf][row * STR + c * 8]);
      async_cp16(Bw + (size_t)(bn + row) * K + kb + c * 8,
                 &Bs[buf][row * STR + c * 8]);
    }
  };

  stage(0, 0);
  async_wait0();
  __syncthreads();

  int buf = 0;
  for (int kb = 0; kb < K; kb += 32) {
    if (kb + 32 < K) stage(buf ^ 1, kb + 32);   // overlap copy with compute

    v16bf af[2];
#pragma unroll
    for (int mi = 0; mi < 2; ++mi) {        // A frag: ISA A-layout 16x32
      v8u u;
      int row = wm + mi * 16 + r;
#pragma unroll
      for (int v = 0; v < 8; ++v) {
        int k0 = (v < 4) ? (hi * 8 + 2 * v) : (16 + hi * 8 + 2 * (v - 4));
        u[v] = *(const unsigned int*)&As[buf][row * STR + k0];
      }
      af[mi] = __builtin_bit_cast(v16bf, u);
    }
    v16bf bfv[4];
#pragma unroll
    for (int ni = 0; ni < 4; ++ni) {        // B frag: col per lane, K packed
      v8u u;
      int row = wn + ni * 16 + r;
      int kh = hi * 8;
#pragma unroll
      for (int j = 0; j < 8; ++j)
        u[j] = *(const unsigned int*)&Bs[buf][row * STR + ((j < 4) ? (kh + 2 * j) : (16 + kh + 2 * (j - 4)))];
      bfv[ni] = __builtin_bit_cast(v16bf, u);
    }
#pragma unroll
    for (int mi = 0; mi < 2; ++mi)
#pragma unroll
      for (int ni = 0; ni < 4; ++ni)
        acc[mi][ni] = __builtin_amdgcn_wmma_f32_16x16x32_bf16(
            false, af[mi], false, bfv[ni], (short)0, acc[mi][ni], false, false);

    async_wait0();
    __syncthreads();
    buf ^= 1;
  }

#pragma unroll
  for (int mi = 0; mi < 2; ++mi)
#pragma unroll
    for (int ni = 0; ni < 4; ++ni)
#pragma unroll
      for (int v = 0; v < 8; ++v) {
        int row = bm + wm + mi * 16 + hi * 8 + v;
        int col = bn + wn + ni * 16 + r;
        C[(size_t)row * N + col] = acc[mi][ni][v];
      }
}

// ---------------------------------------------------------------------------
// Per-head LayerNorm + RoPE + cache scatter. One wave per (token, head).
// heads 0..31 -> Q bf16; 32..39 -> K (fp32 cache + bf16 ws); 40..47 -> V same.
// ---------------------------------------------------------------------------
__global__ __launch_bounds__(256) void lnrope_kernel(
    const float* __restrict__ qkv, const float* __restrict__ cosT,
    const float* __restrict__ sinT, const float* __restrict__ qw,
    const float* __restrict__ kw, const long long* __restrict__ slots,
    unsigned short* __restrict__ q_post, float* __restrict__ k_cache,
    float* __restrict__ v_cache, unsigned short* __restrict__ k_bf,
    unsigned short* __restrict__ v_bf) {
  int wid  = blockIdx.x * 8 + (threadIdx.x >> 5);
  int lane = threadIdx.x & 31;
  int t = wid / 48;
  int h = wid % 48;
  if (t >= TT) return;
  const float* x = qkv + (size_t)t * NQKV + h * DH;
  float v[4];
#pragma unroll
  for (int j = 0; j < 4; ++j) v[j] = x[lane + 32 * j];
  if (h < 40) {
    float s = v[0] + v[1] + v[2] + v[3];
#pragma unroll
    for (int off = 16; off; off >>= 1) s += __shfl_xor(s, off, 32);
    float mu = s * (1.f / 128.f);
    float q2 = 0.f;
#pragma unroll
    for (int j = 0; j < 4; ++j) { v[j] -= mu; q2 += v[j] * v[j]; }
#pragma unroll
    for (int off = 16; off; off >>= 1) q2 += __shfl_xor(q2, off, 32);
    float inv = rsqrtf(q2 * (1.f / 128.f) + 1e-5f);
    const float* w = (h < 32) ? (qw + h * DH) : (kw + (h - 32) * DH);
#pragma unroll
    for (int j = 0; j < 4; ++j) v[j] = w[lane + 32 * j] * v[j] * inv;
    float c0 = cosT[t * 64 + lane],      s0 = sinT[t * 64 + lane];
    float c1 = cosT[t * 64 + lane + 32], s1 = sinT[t * 64 + lane + 32];
    float y0 = v[0] * c0 - v[2] * s0;
    float y2 = v[2] * c0 + v[0] * s0;
    float y1 = v[1] * c1 - v[3] * s1;
    float y3 = v[3] * c1 + v[1] * s1;
    v[0] = y0; v[1] = y1; v[2] = y2; v[3] = y3;
  }
  if (h < 32) {
    unsigned short* d = q_post + ((size_t)t * HQ + h) * DH;
#pragma unroll
    for (int j = 0; j < 4; ++j) d[lane + 32 * j] = f2bf(v[j]);
  } else {
    long long slot = slots[t];
    int hh = (h < 40) ? h - 32 : h - 40;
    size_t off = ((size_t)slot * HKV + hh) * DH;
    float* d = (h < 40 ? k_cache : v_cache) + off;
    unsigned short* dbf = (h < 40 ? k_bf : v_bf) + off;
#pragma unroll
    for (int j = 0; j < 4; ++j) {
      d[lane + 32 * j]   = v[j];
      dbf[lane + 32 * j] = f2bf(v[j]);
    }
  }
}

// ---------------------------------------------------------------------------
// Flash attention: block = 4 waves x 16 queries for one (b, h). bf16 K/V
// double-buffered in LDS via async copies; online softmax in registers.
// ---------------------------------------------------------------------------
__global__ __launch_bounds__(128) void attn_kernel(
    const unsigned short* __restrict__ q_post,
    const unsigned short* __restrict__ k_bf,
    const unsigned short* __restrict__ v_bf,
    unsigned short* __restrict__ attn_out) {
  const int KSTR = 136, PSTR = 34;          // 272B K/V rows (16B aligned)
  __shared__ unsigned short Kt[2][32 * 136];
  __shared__ unsigned short Vt[2][32 * 136];
  __shared__ unsigned short Pw[4 * 16 * 34];
  const int tid = threadIdx.x, lane = tid & 31, wave = tid >> 5;
  const int bh = blockIdx.y;
  const int b = bh >> 5, h = bh & 31;
  const int hk = h >> 2;                    // G = 4
  const int q0 = blockIdx.x * 64;
  const int qbase = q0 + wave * 16;
  const int r = lane & 15, hi = lane >> 4;
  const float scale = 0.08838834764831845f; // D^-0.5

  // Q fragments: 4 chunks of K=32 covering D=128 (A-layout)
  v16bf qf[4];
  {
    int trow = b * SS + qbase + r;
    const unsigned short* qp = q_post + ((size_t)trow * HQ + h) * DH;
#pragma unroll
    for (int c = 0; c < 4; ++c) {
      v8u u;
#pragma unroll
      for (int v = 0; v < 8; ++v) {
        int k0 = (v < 4) ? (hi * 8 + 2 * v) : (16 + hi * 8 + 2 * (v - 4));
        u[v] = *(const unsigned int*)(qp + 32 * c + k0);
      }
      qf[c] = __builtin_bit_cast(v16bf, u);
    }
  }

  v8f o[8];
#pragma unroll
  for (int i = 0; i < 8; ++i) o[i] = (v8f)0.f;
  float m_i[8], l_i[8];
#pragma unroll
  for (int v = 0; v < 8; ++v) { m_i[v] = -1e30f; l_i[v] = 0.f; }

  auto stage = [&](int buf, int kbase) {
#pragma unroll
    for (int it = 0; it < 4; ++it) {
      int idx = tid + it * 128;             // 512 B128 chunks per tile
      int key = idx >> 4, c = idx & 15;
      size_t off = ((size_t)(b * SS + kbase + key) * HKV + hk) * DH + c * 8;
      async_cp16(k_bf + off, &Kt[buf][key * KSTR + c * 8]);
      async_cp16(v_bf + off, &Vt[buf][key * KSTR + c * 8]);
    }
  };

  const int nkb = q0 / 32 + 2;              // causal: keys <= q0+63
  stage(0, 0);
  async_wait0();
  __syncthreads();

  int buf = 0;
  for (int kb = 0; kb < nkb; ++kb) {
    const int kbase = kb * 32;
    if (kb + 1 < nkb) stage(buf ^ 1, kbase + 32);

    if (kbase <= qbase + 15) {
      // scores: Q(16x128) . K^T -> two 16x16 tiles
      v8f sc[2];
#pragma unroll
      for (int nt = 0; nt < 2; ++nt) {
        v8f s = (v8f)0.f;
#pragma unroll
        for (int c = 0; c < 4; ++c) {
          v8u u;
          int key = nt * 16 + r;
          int kh = hi * 8;
#pragma unroll
          for (int j = 0; j < 8; ++j)
            u[j] = *(const unsigned int*)&Kt[buf][key * KSTR + 32 * c + ((j < 4) ? (kh + 2 * j) : (16 + kh + 2 * (j - 4)))];
          v16bf bk = __builtin_bit_cast(v16bf, u);
          s = __builtin_amdgcn_wmma_f32_16x16x32_bf16(false, qf[c], false, bk,
                                                      (short)0, s, false, false);
        }
        sc[nt] = s;
      }
      // causal mask + online softmax (row = hi*8+v, col = kbase + nt*16 + r)
#pragma unroll
      for (int v = 0; v < 8; ++v) {
        int q = qbase + hi * 8 + v;
        float s0 = sc[0][v] * scale, s1 = sc[1][v] * scale;
        if (kbase + r > q)      s0 = -1e30f;
        if (kbase + 16 + r > q) s1 = -1e30f;
        float t = fmaxf(s0, s1);
#pragma unroll
        for (int off = 8; off; off >>= 1) t = fmaxf(t, __shfl_xor(t, off, 32));
        float nm = fmaxf(m_i[v], t);
        float p0 = __expf(s0 - nm), p1 = __expf(s1 - nm);
        float alpha = __expf(m_i[v] - nm);
        float rs = p0 + p1;
#pragma unroll
        for (int off = 8; off; off >>= 1) rs += __shfl_xor(rs, off, 32);
        l_i[v] = l_i[v] * alpha + rs;
        m_i[v] = nm;
#pragma unroll
        for (int dt = 0; dt < 8; ++dt) o[dt][v] *= alpha;
        int row = hi * 8 + v;
        Pw[(wave * 16 + row) * PSTR + r]      = f2bf(p0);
        Pw[(wave * 16 + row) * PSTR + 16 + r] = f2bf(p1);
      }
      // P(16x32) . V(32x128): rebuild P as A-fragment from per-wave LDS
      v16bf pf;
      {
        v8u u;
#pragma unroll
        for (int v = 0; v < 8; ++v) {
          int k0 = (v < 4) ? (hi * 8 + 2 * v) : (16 + hi * 8 + 2 * (v - 4));
          u[v] = *(const unsigned int*)&Pw[(wave * 16 + r) * PSTR + k0];
        }
        pf = __builtin_bit_cast(v16bf, u);
      }
#pragma unroll
      for (int dt = 0; dt < 8; ++dt) {
        v8u u;
        int dcol = dt * 16 + r;
        int kh = hi * 8;
#pragma unroll
        for (int j = 0; j < 8; ++j) {
          int kk = (j < 4) ? (kh + 2 * j) : (16 + kh + 2 * (j - 4));
          unsigned int lo  = Vt[buf][kk * KSTR + dcol];
          unsigned int hi2 = Vt[buf][(kk + 1) * KSTR + dcol];
          u[j] = lo | (hi2 << 16);
        }
        v16bf bv = __builtin_bit_cast(v16bf, u);
        o[dt] = __builtin_amdgcn_wmma_f32_16x16x32_bf16(false, pf, false, bv,
                                                        (short)0, o[dt], false, false);
      }
    }
    async_wait0();
    __syncthreads();
    buf ^= 1;
  }

  // epilogue: normalize rows, write bf16 attn (T, H*D)
#pragma unroll
  for (int v = 0; v < 8; ++v) {
    float inv = 1.f / l_i[v];
    int trow = b * SS + qbase + hi * 8 + v;
    unsigned short* dp = attn_out + ((size_t)trow * HQ + h) * DH;
#pragma unroll
    for (int dt = 0; dt < 8; ++dt) dp[dt * 16 + r] = f2bf(o[dt][v] * inv);
  }
}

// ---------------------------------------------------------------------------
extern "C" void kernel_launch(void* const* d_in, const int* in_sizes, int n_in,
                              void* d_out, int out_size, void* d_ws,
                              size_t ws_size, hipStream_t stream) {
  (void)in_sizes; (void)n_in; (void)out_size; (void)ws_size;
  const float* hidden     = (const float*)d_in[0];
  const float* cosT       = (const float*)d_in[1];
  const float* sinT       = (const float*)d_in[2];
  const float* qkv_w      = (const float*)d_in[3];
  const float* q_norm_w   = (const float*)d_in[4];
  const float* k_norm_w   = (const float*)d_in[5];
  const float* o_w        = (const float*)d_in[6];
  const long long* slots  = (const long long*)d_in[9];

  float* out     = (float*)d_out;
  float* k_cache = out + (size_t)TT * HID;
  float* v_cache = k_cache + (size_t)TT * HKV * DH;

  // workspace layout
  float* qkv                = (float*)d_ws;                         // T*6144 f32
  unsigned short* hidden_bf = (unsigned short*)(qkv + (size_t)TT * NQKV);
  unsigned short* qkvw_bf   = hidden_bf + (size_t)TT * HID;
  unsigned short* ow_bf     = qkvw_bf + (size_t)NQKV * HID;
  unsigned short* q_post    = ow_bf + (size_t)HID * HID;
  unsigned short* attn      = q_post + (size_t)TT * HQ * DH;
  unsigned short* k_bf      = attn + (size_t)TT * HQ * DH;
  unsigned short* v_bf      = k_bf + (size_t)TT * HKV * DH;

  // 0) one-shot bf16 conversions (bandwidth-bound)
  cvt_bf16_kernel<<<(TT * HID / 4) / 256, 256, 0, stream>>>(
      hidden, hidden_bf, (long long)TT * HID / 4);
  cvt_bf16_kernel<<<((size_t)NQKV * HID / 4) / 256, 256, 0, stream>>>(
      qkv_w, qkvw_bf, (long long)NQKV * HID / 4);
  cvt_bf16_kernel<<<((size_t)HID * HID / 4) / 256, 256, 0, stream>>>(
      o_w, ow_bf, (long long)HID * HID / 4);

  // 1) QKV projection: (T,4096) x (6144,4096)^T
  gemm_bf16_kernel<<<dim3(NQKV / 128, TT / 128), 256, 0, stream>>>(
      hidden_bf, qkvw_bf, qkv, TT, NQKV, HID);
  // 2) per-head LN + RoPE + cache scatter (+ bf16 K/V copies)
  lnrope_kernel<<<(TT * 48) / 8, 256, 0, stream>>>(
      qkv, cosT, sinT, q_norm_w, k_norm_w, slots, q_post, k_cache, v_cache,
      k_bf, v_bf);
  // 3) flash attention
  attn_kernel<<<dim3(SS / 64, BB * HQ), 128, 0, stream>>>(
      q_post, k_bf, v_bf, attn);
  // 4) O projection: (T,4096) x (4096,4096)^T
  gemm_bf16_kernel<<<dim3(HID / 128, TT / 128), 256, 0, stream>>>(
      attn, ow_bf, out, TT, HID, HID);
}
